// Fine_82154134438334
// MI455X (gfx1250) — matrix operand order repaired and checked by reading, not compiled, not run on hardware
//
#include <hip/hip_runtime.h>
#include <hip/hip_bf16.h>
#include <math.h>

typedef __attribute__((ext_vector_type(16))) _Float16 v16h;
typedef __attribute__((ext_vector_type(8)))  float    v8f;

#define NB    2
#define NPTS  2048
#define KNBR  32
#define NKCOL (KNBR*NPTS)      /* 65536 columns for the local MLP */
#define CF    512
#define C5    2560
#define NTOP1 1536
#define NTOP2 614
#define RAD2  0.09f

// ---------------------------------------------------------------------------
// Generic WMMA GEMM:  C[M,N] = (A @ B) + bias
//   A: f32, [M,K] row-major (TA=0) or [K,M] row-major (TA=1)
//   B: f32, [K,N] row-major (TB=0) or [N,K] row-major (TB=1)
// f32 -> f16 while staging into LDS *already in WMMA fragment layout*, so each
// wave's fragment load is a contiguous 32-byte LDS read (ds_load_b128 x2).
// Block tile 64x64, 8 waves; each wave: one A fragment, two 16x16 WMMA tiles.
// GUARD=false: fully unguarded straight-line inner loop (dims are multiples of
// the tile); GUARD=true: per-element zero-padded edges (used for K=13 layer).
// ---------------------------------------------------------------------------
#define TM 64
#define TN 64
#define TK 32

template <int TA, int TB, bool GUARD>
__launch_bounds__(256)
__global__ void k_gemm_t(const float* __restrict__ A, const float* __restrict__ Bm,
                         const float* __restrict__ bias, float* __restrict__ C,
                         int M, int N, int Kd,
                         long sA, long sB, long sC, int hasBias)
{
  // fragment-layout LDS: [tile][lane] -> 16 halfs, contiguous per lane
  __shared__ v16h AsF[4][32];   // 4 M-tiles of 16 rows
  __shared__ v16h BsF[4][32];   // 4 N-tiles of 16 cols

  const int bz = blockIdx.z;
  const float* Ab = A  + (long)bz * sA;
  const float* Bb = Bm + (long)bz * sB;
  float*       Cb = C  + (long)bz * sC;

  const int m0 = blockIdx.y * TM;
  const int n0 = blockIdx.x * TN;
  const int tid  = threadIdx.x;
  const int wave = tid >> 5;
  const int lane = tid & 31;          // wave32
  const int wtm  = wave & 3;          // M-tile of this wave
  const int wtn  = (wave >> 2) * 2;   // first of two N-tiles of this wave

  // staging coordinates (constant per thread); inverse fragment mappings:
  //   A(16x32): lane = (row&15)+16*((k>>3)&1), h = (k&7)+8*((k>>4)&1)
  //   B(32x16): lane = (n&15)+16*((k>>3)&1),   h = (k&7)+8*((k>>4)&1)
  const int arow = tid >> 2;                 // 0..63
  const int akc0 = (tid & 3) * 8;            // 0,8,16,24
  const int aln  = (arow & 15) + (((akc0 >> 3) & 1) << 4);
  const int ahh  = ((akc0 >> 4) & 1) << 3;
  _Float16* adst = (_Float16*)&AsF[arow >> 4][aln] + ahh;

  const int bnc  = tid & 63;                 // 0..63
  const int bkr0 = (tid >> 6) * 8;           // 0,8,16,24
  const int bln  = (bnc & 15) + (((bkr0 >> 3) & 1) << 4);
  const int bhh  = ((bkr0 >> 4) & 1) << 3;
  _Float16* bdst = (_Float16*)&BsF[bnc >> 4][bln] + bhh;

  v8f acc0 = {}, acc1 = {};

  for (int k0 = 0; k0 < Kd; k0 += TK) {
    // next-tile prefetch (global_prefetch_b8; speculative, safe if OOB)
    if (TA == 0)
      __builtin_prefetch(&Ab[(long)(m0 + arow) * Kd + k0 + TK], 0, 1);
    if (TB == 0)
      __builtin_prefetch(&Bb[(long)(k0 + TK + bkr0) * N + n0 + bnc], 0, 1);

    if (GUARD) {
      const int gm = m0 + arow;
#pragma unroll
      for (int u = 0; u < 8; ++u) {
        int gk = k0 + akc0 + u;
        float v = 0.f;
        if (gm < M && gk < Kd)
          v = TA ? Ab[(long)gk * M + gm] : Ab[(long)gm * Kd + gk];
        adst[u] = (_Float16)v;
      }
      const int gn = n0 + bnc;
#pragma unroll
      for (int u = 0; u < 8; ++u) {
        int gk = k0 + bkr0 + u;
        float v = 0.f;
        if (gn < N && gk < Kd)
          v = TB ? Bb[(long)gn * Kd + gk] : Bb[(long)gk * N + gn];
        bdst[u] = (_Float16)v;
      }
    } else {
      const int gm = m0 + arow;
#pragma unroll
      for (int u = 0; u < 8; ++u) {
        int gk = k0 + akc0 + u;
        float v = TA ? Ab[(long)gk * M + gm] : Ab[(long)gm * Kd + gk];
        adst[u] = (_Float16)v;
      }
      const int gn = n0 + bnc;
#pragma unroll
      for (int u = 0; u < 8; ++u) {
        int gk = k0 + bkr0 + u;
        float v = TB ? Bb[(long)gn * Kd + gk] : Bb[(long)gk * N + gn];
        bdst[u] = (_Float16)v;
      }
    }
    __syncthreads();

    // contiguous 32B fragment loads (ds_load_b128 pairs)
    v16h af = AsF[wtm][lane];
    v16h b0 = BsF[wtn][lane];
    v16h b1 = BsF[wtn + 1][lane];

    acc0 = __builtin_amdgcn_wmma_f32_16x16x32_f16(false, af, false, b0,
                                                  (short)0, acc0, false, false);
    acc1 = __builtin_amdgcn_wmma_f32_16x16x32_f16(false, af, false, b1,
                                                  (short)0, acc1, false, false);
    __syncthreads();
  }

  // C/D layout: lanes 0-15 -> N=lane, M=r ; lanes 16-31 -> N=lane-16, M=r+8
  const int nn   = lane & 15;
  const int moff = (lane >> 4) * 8;
#pragma unroll
  for (int r = 0; r < 8; ++r) {
    int gm = m0 + wtm * 16 + moff + r;
    int gn0 = n0 + wtn * 16 + nn;
    int gn1 = gn0 + 16;
    if (GUARD) {
      if (gm >= M) continue;
      float bv = hasBias ? bias[gm] : 0.f;
      if (gn0 < N) Cb[(long)gm * N + gn0] = acc0[r] + bv;
      if (gn1 < N) Cb[(long)gm * N + gn1] = acc1[r] + bv;
    } else {
      float bv = hasBias ? bias[gm] : 0.f;
      Cb[(long)gm * N + gn0] = acc0[r] + bv;
      Cb[(long)gm * N + gn1] = acc1[r] + bv;
    }
  }
}

// ---------------------------------------------------------------------------
// Ball query: first K ascending neighbors with d2 <= r^2, fallback = first.
// ---------------------------------------------------------------------------
__global__ void k_ball(const float* __restrict__ xyz, int* __restrict__ nbr)
{
  int t = blockIdx.x * blockDim.x + threadIdx.x;
  if (t >= NB * NPTS) return;
  int b = t / NPTS, n = t % NPTS;
  const float* P = xyz + (long)b * NPTS * 3;
  float px = P[3 * n], py = P[3 * n + 1], pz = P[3 * n + 2];
  int* out = nbr + (long)t * KNBR;
  int cnt = 0, first = -1;
  for (int m = 0; m < NPTS && cnt < KNBR; ++m) {
    float dx = P[3 * m] - px, dy = P[3 * m + 1] - py, dz = P[3 * m + 2] - pz;
    float d2 = dx * dx + dy * dy + dz * dz;
    if (d2 <= RAD2) { if (first < 0) first = m; out[cnt++] = m; }
  }
  if (first < 0) first = n;
  for (; cnt < KNBR; ++cnt) out[cnt] = first;
}

__device__ __forceinline__ float angf(float ax, float ay, float az,
                                      float bx, float by, float bz)
{
  float cx = ay * bz - az * by;
  float cy = az * bx - ax * bz;
  float cz = ax * by - ay * bx;
  float cn = sqrtf(cx * cx + cy * cy + cz * cz);
  float dp = ax * bx + ay * by + az * bz;
  return atan2f(cn, dp);
}

// 13-channel PPF feature tensor X[b][c][k*NPTS + n]
__global__ void k_feats(const float* __restrict__ xyz, const float* __restrict__ nrm,
                        const int* __restrict__ nbr,
                        const float* __restrict__ wa, const float* __restrict__ wb,
                        const float* __restrict__ wc, float* __restrict__ X)
{
  int t = blockIdx.x * blockDim.x + threadIdx.x;
  if (t >= NB * KNBR * NPTS) return;
  int n = t % NPTS;
  int k = (t / NPTS) % KNBR;
  int b = t / (NPTS * KNBR);
  const float* P = xyz + (long)b * NPTS * 3;
  const float* Q = nrm + (long)b * NPTS * 3;
  int j = nbr[((long)b * NPTS + n) * KNBR + k];
  float px = P[3 * n], py = P[3 * n + 1], pz = P[3 * n + 2];
  float qx = P[3 * j], qy = P[3 * j + 1], qz = P[3 * j + 2];
  float rx = qx - px, ry = qy - py, rz = qz - pz;
  float d  = sqrtf(rx * rx + ry * ry + rz * rz);
  float n0x = Q[3 * n], n0y = Q[3 * n + 1], n0z = Q[3 * n + 2];
  float n1x = Q[3 * j], n1y = Q[3 * j + 1], n1z = Q[3 * j + 2];
  float a = *wa, bb = *wb, cc = *wc;
  long col  = (long)k * NPTS + n;
  float* Xb = X + (long)b * 13 * NKCOL;
  Xb[0L * NKCOL + col]  = a * px;
  Xb[1L * NKCOL + col]  = a * py;
  Xb[2L * NKCOL + col]  = a * pz;
  Xb[3L * NKCOL + col]  = a * qx;
  Xb[4L * NKCOL + col]  = a * qy;
  Xb[5L * NKCOL + col]  = a * qz;
  Xb[6L * NKCOL + col]  = bb * rx;
  Xb[7L * NKCOL + col]  = bb * ry;
  Xb[8L * NKCOL + col]  = bb * rz;
  Xb[9L * NKCOL + col]  = bb * d;
  Xb[10L * NKCOL + col] = cc * angf(n0x, n0y, n0z, rx, ry, rz);
  Xb[11L * NKCOL + col] = cc * angf(n1x, n1y, n1z, rx, ry, rz);
  Xb[12L * NKCOL + col] = cc * angf(n0x, n0y, n0z, n1x, n1y, n1z);
}

// ---------------------------------------------------------------------------
// Group norm (channels-contiguous groups): stats then apply(+act)
// ---------------------------------------------------------------------------
__global__ void k_gn_stats(const float* __restrict__ x, float* __restrict__ mean,
                           float* __restrict__ rstd, int C, long ncols, int groups,
                           float eps)
{
  __shared__ float s0[256];
  __shared__ float s1[256];
  int g = blockIdx.x, b = blockIdx.y;
  int cpg = C / groups;
  const float* xb = x + (long)b * C * ncols + (long)g * cpg * ncols;
  long total = (long)cpg * ncols;
  float s = 0.f, q = 0.f;
  for (long i = threadIdx.x; i < total; i += blockDim.x) {
    float v = xb[i]; s += v; q += v * v;
  }
  s0[threadIdx.x] = s; s1[threadIdx.x] = q;
  __syncthreads();
  for (int o = 128; o > 0; o >>= 1) {
    if ((int)threadIdx.x < o) { s0[threadIdx.x] += s0[threadIdx.x + o]; s1[threadIdx.x] += s1[threadIdx.x + o]; }
    __syncthreads();
  }
  if (threadIdx.x == 0) {
    float m = s0[0] / (float)total;
    float v = s1[0] / (float)total - m * m;
    mean[b * groups + g] = m;
    rstd[b * groups + g] = rsqrtf(v + eps);
  }
}

__global__ void k_gn_apply(float* __restrict__ x, const float* __restrict__ gamma,
                           const float* __restrict__ beta, const float* __restrict__ mean,
                           const float* __restrict__ rstd, int C, long ncols, int groups,
                           int act, float slope, long total)
{
  int cpg = C / groups;
  for (long i = blockIdx.x * (long)blockDim.x + threadIdx.x; i < total;
       i += (long)gridDim.x * blockDim.x) {
    long b  = i / ((long)C * ncols);
    long rm = i - b * (long)C * ncols;
    int  c  = (int)(rm / ncols);
    int  g  = c / cpg;
    float v = (x[i] - mean[b * groups + g]) * rstd[b * groups + g] * gamma[c] + beta[c];
    if (act == 1)      v = fmaxf(v, 0.f);
    else if (act == 2) v = (v > 0.f) ? v : slope * v;
    x[i] = v;
  }
}

// max over K neighbors:  out = max_k x ; outgn = relu(max_k x)
__global__ void k_maxk(const float* __restrict__ x, float* __restrict__ out, long outB,
                       float* __restrict__ outgn, long ognB)
{
  long total = (long)NB * CF * NPTS;
  for (long i = blockIdx.x * (long)blockDim.x + threadIdx.x; i < total;
       i += (long)gridDim.x * blockDim.x) {
    long b  = i / ((long)CF * NPTS);
    long rm = i - b * (long)CF * NPTS;
    int  c  = (int)(rm / NPTS);
    int  n  = (int)(rm % NPTS);
    const float* xp = x + ((long)b * CF + c) * (long)NKCOL + n;
    float mx = -1e30f;
    for (int k = 0; k < KNBR; ++k) mx = fmaxf(mx, xp[(long)k * NPTS]);
    out[b * outB + (long)c * NPTS + n]    = mx;
    outgn[b * ognB + (long)c * NPTS + n]  = fmaxf(mx, 0.f);
  }
}

// row softmax over Ncol
__global__ void k_softmax(float* __restrict__ a, int Nrow, int Ncol)
{
  __shared__ float sh[256];
  int b = blockIdx.y, n = blockIdx.x;
  float* r = a + ((long)b * Nrow + n) * Ncol;
  float mx = -1e30f;
  for (int m = threadIdx.x; m < Ncol; m += 256) mx = fmaxf(mx, r[m]);
  sh[threadIdx.x] = mx; __syncthreads();
  for (int o = 128; o > 0; o >>= 1) {
    if ((int)threadIdx.x < o) sh[threadIdx.x] = fmaxf(sh[threadIdx.x], sh[threadIdx.x + o]);
    __syncthreads();
  }
  mx = sh[0]; __syncthreads();
  float s = 0.f;
  for (int m = threadIdx.x; m < Ncol; m += 256) { float e = __expf(r[m] - mx); r[m] = e; s += e; }
  sh[threadIdx.x] = s; __syncthreads();
  for (int o = 128; o > 0; o >>= 1) {
    if ((int)threadIdx.x < o) sh[threadIdx.x] += sh[threadIdx.x + o];
    __syncthreads();
  }
  float inv = 1.f / sh[0];
  for (int m = threadIdx.x; m < Ncol; m += 256) r[m] *= inv;
}

// column normalize: a[:,m] /= (1e-9 + sum_n a[n,m])
__global__ void k_colnorm(float* __restrict__ a, int Nrow, int Ncol)
{
  __shared__ float sh[256];
  int b = blockIdx.y, m = blockIdx.x;
  float* ab = a + (long)b * Nrow * Ncol;
  float s = 0.f;
  for (int n = threadIdx.x; n < Nrow; n += 256) s += ab[(long)n * Ncol + m];
  sh[threadIdx.x] = s; __syncthreads();
  for (int o = 128; o > 0; o >>= 1) {
    if ((int)threadIdx.x < o) sh[threadIdx.x] += sh[threadIdx.x + o];
    __syncthreads();
  }
  float inv = 1.f / (1e-9f + sh[0]);
  for (int n = threadIdx.x; n < Nrow; n += 256) ab[(long)n * Ncol + m] *= inv;
}

__global__ void k_ewsub(const float* __restrict__ a, const float* __restrict__ b,
                        float* __restrict__ y, long total)
{
  for (long i = blockIdx.x * (long)blockDim.x + threadIdx.x; i < total;
       i += (long)gridDim.x * blockDim.x) y[i] = a[i] - b[i];
}

__global__ void k_ewadd(float* __restrict__ a, const float* __restrict__ b, long total)
{
  for (long i = blockIdx.x * (long)blockDim.x + threadIdx.x; i < total;
       i += (long)gridDim.x * blockDim.x) a[i] += b[i];
}

__global__ void k_copy_slot(const float* __restrict__ x, float* __restrict__ dst,
                            long perB, long dstB, long dstOff, long total)
{
  for (long i = blockIdx.x * (long)blockDim.x + threadIdx.x; i < total;
       i += (long)gridDim.x * blockDim.x) {
    long b = i / perB, r = i - b * perB;
    dst[b * dstB + dstOff + r] = x[i];
  }
}

// transpose [C5][NPTS] -> [NPTS][C5] with row L2-normalization
__global__ void k_transnorm(const float* __restrict__ f, float* __restrict__ o)
{
  __shared__ float sh[256];
  int b = blockIdx.y, n = blockIdx.x;
  const float* fb = f + (long)b * C5 * NPTS;
  float s = 0.f;
  for (int c = threadIdx.x; c < C5; c += 256) { float v = fb[(long)c * NPTS + n]; s += v * v; }
  sh[threadIdx.x] = s; __syncthreads();
  for (int w = 128; w > 0; w >>= 1) {
    if ((int)threadIdx.x < w) sh[threadIdx.x] += sh[threadIdx.x + w];
    __syncthreads();
  }
  float inv = 1.f / (sqrtf(sh[0]) + 1e-8f);
  for (int c = threadIdx.x; c < C5; c += 256)
    o[((long)b * NPTS + n) * C5 + c] = fb[(long)c * NPTS + n] * inv;
}

// descending bitonic argsort (npad = 2048, power of two)
__launch_bounds__(1024)
__global__ void k_argsort(const float* __restrict__ vals, float* __restrict__ sortedv,
                          int* __restrict__ order, int n, int npad)
{
  __shared__ float sk[2048];
  __shared__ int   si[2048];
  int b = blockIdx.x;
  const float* v = vals + (long)b * n;
  for (int i = threadIdx.x; i < npad; i += blockDim.x) { sk[i] = (i < n) ? v[i] : -1e30f; si[i] = i; }
  __syncthreads();
  for (int ks = 2; ks <= npad; ks <<= 1) {
    for (int j = ks >> 1; j > 0; j >>= 1) {
      for (int i = threadIdx.x; i < npad; i += blockDim.x) {
        int ixj = i ^ j;
        if (ixj > i) {
          bool up = ((i & ks) == 0);
          float a = sk[i], c = sk[ixj];
          bool sw = up ? (a < c) : (a > c);   // descending overall
          if (sw) { sk[i] = c; sk[ixj] = a; int t = si[i]; si[i] = si[ixj]; si[ixj] = t; }
        }
      }
      __syncthreads();
    }
  }
  for (int i = threadIdx.x; i < n; i += blockDim.x) {
    sortedv[(long)b * n + i] = sk[i];
    order[(long)b * n + i]   = si[i];
  }
}

__global__ void k_gather_feat(const float* __restrict__ ftn, const int* __restrict__ ord,
                              float* __restrict__ feat1)
{
  long total = (long)NB * NTOP1 * C5;
  for (long i = blockIdx.x * (long)blockDim.x + threadIdx.x; i < total;
       i += (long)gridDim.x * blockDim.x) {
    long b  = i / ((long)NTOP1 * C5);
    long rm = i - b * (long)NTOP1 * C5;
    int  rw = (int)(rm / C5);
    int  c  = (int)(rm % C5);
    int  s  = ord[b * NPTS + rw];
    feat1[i] = ftn[((long)b * NPTS + s) * C5 + c];
  }
}

__global__ void k_gather_pts(const float* __restrict__ pts, const int* __restrict__ ord,
                             float* __restrict__ pts1)
{
  int t = blockIdx.x * blockDim.x + threadIdx.x;
  if (t >= NB * NTOP1 * 3) return;
  int b  = t / (NTOP1 * 3);
  int rm = t % (NTOP1 * 3);
  int rw = rm / 3, d = rm % 3;
  int s  = ord[b * NPTS + rw];
  pts1[t] = pts[((long)b * NPTS + s) * 3 + d];
}

__global__ void k_rowmax(const float* __restrict__ sim, float* __restrict__ smax)
{
  __shared__ float sh[256];
  int b = blockIdx.y, n = blockIdx.x;
  const float* r = sim + ((long)b * NTOP1 + n) * NTOP1;
  float mx = -1e30f;
  for (int m = threadIdx.x; m < NTOP1; m += 256) mx = fmaxf(mx, r[m]);
  sh[threadIdx.x] = mx; __syncthreads();
  for (int o = 128; o > 0; o >>= 1) {
    if ((int)threadIdx.x < o) sh[threadIdx.x] = fmaxf(sh[threadIdx.x], sh[threadIdx.x + o]);
    __syncthreads();
  }
  if (threadIdx.x == 0) smax[(long)b * NTOP1 + n] = sh[0];
}

// top-3 weighted correspondences + final output assembly
__global__ void k_finalize(const float* __restrict__ sim, const int* __restrict__ smord,
                           const float* __restrict__ src1, const float* __restrict__ tgt1,
                           const float* __restrict__ xols, float* __restrict__ out)
{
  int t = blockIdx.x * blockDim.x + threadIdx.x;
  if (t >= NB * NTOP2) return;
  int b = t / NTOP2, i = t % NTOP2;
  int idx = smord[b * NTOP1 + i];

  float* o_src2 = out;
  float* o_corr = out + (size_t)NB * NTOP2 * 3;
  float* o_icp  = out + (size_t)2 * NB * NTOP2 * 3;
  float* o_inds = o_icp + (size_t)NB * NTOP2;

  const float* s1 = src1 + ((long)b * NTOP1 + idx) * 3;
  o_src2[(size_t)t * 3 + 0] = s1[0];
  o_src2[(size_t)t * 3 + 1] = s1[1];
  o_src2[(size_t)t * 3 + 2] = s1[2];
  o_icp[t]  = xols[(long)b * NPTS + idx];
  o_inds[t] = (float)idx;

  const float* row = sim + ((long)b * NTOP1 + idx) * NTOP1;
  float v0 = -1e30f, v1 = -1e30f, v2 = -1e30f;
  int   j0 = 0, j1 = 0, j2 = 0;
  for (int m = 0; m < NTOP1; ++m) {
    float v = row[m];
    if (v > v0)      { v2 = v1; j2 = j1; v1 = v0; j1 = j0; v0 = v; j0 = m; }
    else if (v > v1) { v2 = v1; j2 = j1; v1 = v; j1 = m; }
    else if (v > v2) { v2 = v; j2 = m; }
  }
  float inv = 1.f / (v0 + v1 + v2 + 1e-8f);
  const float* T = tgt1 + (long)b * NTOP1 * 3;
  for (int d = 0; d < 3; ++d)
    o_corr[(size_t)t * 3 + d] =
        (v0 * T[j0 * 3 + d] + v1 * T[j1 * 3 + d] + v2 * T[j2 * 3 + d]) * inv;
}

// ---------------------------------------------------------------------------
// Host orchestration
// ---------------------------------------------------------------------------
struct Scratch {
  int*   nbr;   float* X0;    float* actA;  float* actB;
  float* gmean; float* grstd; float* xcur;  float* catb;
  float* qk;    float* xv;    float* amat;  float* xr;   float* ybuf;
  float* fbuf;  float* ftn;   int*   ord;   float* sortv;
  float* feat1s; float* feat1t; float* pts1s; float* pts1t;
  float* xols;  float* sim;   float* smax;  int* smord;  float* smsort;
};

static inline unsigned nblk(long total)  { return (unsigned)((total + 255) / 256); }
static inline unsigned nblkc(long total) { long b = (total + 255) / 256; if (b > 32768) b = 32768; return (unsigned)b; }

extern "C" void kernel_launch(void* const* d_in, const int* in_sizes, int n_in,
                              void* d_out, int out_size, void* d_ws, size_t ws_size,
                              hipStream_t stream)
{
  (void)in_sizes; (void)n_in; (void)out_size; (void)ws_size;

  const float* src  = (const float*)d_in[0];
  const float* tgt  = (const float*)d_in[1];
  const float* xol  = (const float*)d_in[2];
  const float* yol  = (const float*)d_in[3];
  const float* nsrc = (const float*)d_in[6];
  const float* ntgt = (const float*)d_in[7];
  const float* wfa  = (const float*)d_in[20];
  const float* wfb  = (const float*)d_in[21];
  const float* wfc  = (const float*)d_in[22];

  // bump-allocate workspace
  char* wsb = (char*)d_ws;
  size_t off = 0;
  auto take = [&](size_t bytes) -> void* {
    void* p = wsb + off;
    off += (bytes + 255) & ~(size_t)255;
    return p;
  };
  Scratch S;
  S.nbr    = (int*)  take(sizeof(int)   * (size_t)NB * NPTS * KNBR);
  S.X0     = (float*)take(sizeof(float) * (size_t)NB * 13 * NKCOL);
  S.actA   = (float*)take(sizeof(float) * (size_t)NB * 512 * NKCOL);
  S.actB   = (float*)take(sizeof(float) * (size_t)NB * 512 * NKCOL);
  S.gmean  = (float*)take(sizeof(float) * (size_t)NB * 16);
  S.grstd  = (float*)take(sizeof(float) * (size_t)NB * 16);
  S.xcur   = (float*)take(sizeof(float) * (size_t)NB * CF * NPTS);
  S.catb   = (float*)take(sizeof(float) * (size_t)NB * C5 * NPTS);
  S.qk     = (float*)take(sizeof(float) * (size_t)NB * 128 * NPTS);
  S.xv     = (float*)take(sizeof(float) * (size_t)NB * CF * NPTS);
  S.amat   = (float*)take(sizeof(float) * (size_t)NB * NPTS * NPTS);
  S.xr     = (float*)take(sizeof(float) * (size_t)NB * CF * NPTS);
  S.ybuf   = (float*)take(sizeof(float) * (size_t)NB * CF * NPTS);
  S.fbuf   = (float*)take(sizeof(float) * (size_t)NB * C5 * NPTS);
  S.ftn    = (float*)take(sizeof(float) * (size_t)NB * NPTS * C5);
  S.ord    = (int*)  take(sizeof(int)   * (size_t)NB * NPTS);
  S.sortv  = (float*)take(sizeof(float) * (size_t)NB * NPTS);
  S.feat1s = (float*)take(sizeof(float) * (size_t)NB * NTOP1 * C5);
  S.feat1t = (float*)take(sizeof(float) * (size_t)NB * NTOP1 * C5);
  S.pts1s  = (float*)take(sizeof(float) * (size_t)NB * NTOP1 * 3);
  S.pts1t  = (float*)take(sizeof(float) * (size_t)NB * NTOP1 * 3);
  S.xols   = (float*)take(sizeof(float) * (size_t)NB * NPTS);
  S.sim    = (float*)take(sizeof(float) * (size_t)NB * NTOP1 * NTOP1);
  S.smax   = (float*)take(sizeof(float) * (size_t)NB * NTOP1);
  S.smord  = (int*)  take(sizeof(int)   * (size_t)NB * NTOP1);
  S.smsort = (float*)take(sizeof(float) * (size_t)NB * NTOP1);

  auto gemm = [&](const float* A, long sA, int tA,
                  const float* Bp, long sB, int tB,
                  const float* bias, float* Cp, long sC,
                  int M, int N, int Kd) {
    dim3 g((N + TN - 1) / TN, (M + TM - 1) / TM, NB);
    bool clean = (M % TM == 0) && (N % TN == 0) && (Kd % TK == 0);
    int hb = bias ? 1 : 0;
    if (tA == 0 && tB == 0) {
      if (clean) k_gemm_t<0, 0, false><<<g, 256, 0, stream>>>(A, Bp, bias, Cp, M, N, Kd, sA, sB, sC, hb);
      else       k_gemm_t<0, 0, true ><<<g, 256, 0, stream>>>(A, Bp, bias, Cp, M, N, Kd, sA, sB, sC, hb);
    } else if (tA == 1) {
      if (clean) k_gemm_t<1, 0, false><<<g, 256, 0, stream>>>(A, Bp, bias, Cp, M, N, Kd, sA, sB, sC, hb);
      else       k_gemm_t<1, 0, true ><<<g, 256, 0, stream>>>(A, Bp, bias, Cp, M, N, Kd, sA, sB, sC, hb);
    } else {
      if (clean) k_gemm_t<0, 1, false><<<g, 256, 0, stream>>>(A, Bp, bias, Cp, M, N, Kd, sA, sB, sC, hb);
      else       k_gemm_t<0, 1, true ><<<g, 256, 0, stream>>>(A, Bp, bias, Cp, M, N, Kd, sA, sB, sC, hb);
    }
  };
  auto gnorm = [&](float* x, int C, long ncols, int groups, const float* gamma,
                   const float* beta, int act, float slope) {
    k_gn_stats<<<dim3(groups, NB), 256, 0, stream>>>(x, S.gmean, S.grstd, C, ncols,
                                                     groups, 1e-5f);
    long total = (long)NB * C * ncols;
    k_gn_apply<<<nblkc(total), 256, 0, stream>>>(x, gamma, beta, S.gmean, S.grstd,
                                                 C, ncols, groups, act, slope, total);
  };

  struct Lyr { const float *w, *g, *b; int cin, cout; };
  const Lyr L[4] = {
    {(const float*)d_in[8],  (const float*)d_in[9],  (const float*)d_in[10],  13, 128},
    {(const float*)d_in[11], (const float*)d_in[12], (const float*)d_in[13], 128, 256},
    {(const float*)d_in[14], (const float*)d_in[15], (const float*)d_in[16], 256, 512},
    {(const float*)d_in[17], (const float*)d_in[18], (const float*)d_in[19], 512, 512},
  };

  auto cloud = [&](const float* pts, const float* nrm, const float* ol,
                   float* feat1, float* pts1, float* olSorted) {
    // local features
    k_ball <<<nblk((long)NB * NPTS),        256, 0, stream>>>(pts, S.nbr);
    k_feats<<<nblk((long)NB * NPTS * KNBR), 256, 0, stream>>>(pts, nrm, S.nbr,
                                                              wfa, wfb, wfc, S.X0);
    const float* cur = S.X0;
    long curB = (long)13 * NKCOL;
    float* pp = S.actA; float* qq = S.actB;
    for (int i = 0; i < 4; ++i) {
      gemm(L[i].w, 0, 0, cur, curB, 0, nullptr, pp, (long)L[i].cout * NKCOL,
           L[i].cout, NKCOL, L[i].cin);
      gnorm(pp, L[i].cout, NKCOL, 8, L[i].g, L[i].b, (i < 3) ? 1 : 0, 0.f);
      cur = pp; curB = (long)L[i].cout * NKCOL;
      float* t = pp; pp = qq; qq = t;
    }
    // max over K:  x_ngn -> cat slot 0 ; relu'd -> xcur (attention input)
    k_maxk<<<nblkc((long)NB * CF * NPTS), 256, 0, stream>>>(cur, S.catb,
        (long)C5 * NPTS, S.xcur, (long)CF * NPTS);

    long tot = (long)NB * CF * NPTS;
    for (int p = 0; p < 4; ++p) {
      int base = 23 + 7 * p;
      const float* wqk = (const float*)d_in[base];
      const float* wv  = (const float*)d_in[base + 1];
      const float* bv  = (const float*)d_in[base + 2];
      const float* wt  = (const float*)d_in[base + 3];
      const float* bt  = (const float*)d_in[base + 4];
      const float* gg  = (const float*)d_in[base + 5];
      const float* gb  = (const float*)d_in[base + 6];
      gemm(wqk, 0, 0, S.xcur, (long)CF * NPTS, 0, nullptr, S.qk,
           (long)128 * NPTS, 128, NPTS, CF);
      gemm(wv, 0, 0, S.xcur, (long)CF * NPTS, 0, bv, S.xv,
           (long)CF * NPTS, CF, NPTS, CF);
      // a[n,m] = sum_o q[o,n] q[o,m]
      gemm(S.qk, (long)128 * NPTS, 1, S.qk, (long)128 * NPTS, 0, nullptr, S.amat,
           (long)NPTS * NPTS, NPTS, NPTS, 128);
      k_softmax<<<dim3(NPTS, NB), 256, 0, stream>>>(S.amat, NPTS, NPTS);
      k_colnorm<<<dim3(NPTS, NB), 256, 0, stream>>>(S.amat, NPTS, NPTS);
      gemm(S.xv, (long)CF * NPTS, 0, S.amat, (long)NPTS * NPTS, 0, nullptr, S.xr,
           (long)CF * NPTS, CF, NPTS, NPTS);
      k_ewsub<<<nblkc(tot), 256, 0, stream>>>(S.xcur, S.xr, S.ybuf, tot);
      gemm(wt, 0, 0, S.ybuf, (long)CF * NPTS, 0, bt, S.xv,
           (long)CF * NPTS, CF, NPTS, CF);
      gnorm(S.xv, CF, NPTS, 16, gg, gb, 1, 0.f);
      k_ewadd<<<nblkc(tot), 256, 0, stream>>>(S.xcur, S.xv, tot);
      k_copy_slot<<<nblkc(tot), 256, 0, stream>>>(S.xcur, S.catb, (long)CF * NPTS,
          (long)C5 * NPTS, (long)(CF * (p + 1)) * NPTS, tot);
    }
    // fuse + GN(16) + leaky relu, transpose + L2 normalize
    gemm((const float*)d_in[51], 0, 0, S.catb, (long)C5 * NPTS, 0, nullptr, S.fbuf,
         (long)C5 * NPTS, C5, NPTS, C5);
    gnorm(S.fbuf, C5, NPTS, 16, (const float*)d_in[52], (const float*)d_in[53], 2, 0.2f);
    k_transnorm<<<dim3(NPTS, NB), 256, 0, stream>>>(S.fbuf, S.ftn);
    // argsort overlap scores desc, gather top-N1 features + points
    k_argsort<<<NB, 1024, 0, stream>>>(ol, olSorted, S.ord, NPTS, 2048);
    k_gather_feat<<<nblkc((long)NB * NTOP1 * C5), 256, 0, stream>>>(S.ftn, S.ord, feat1);
    k_gather_pts <<<nblk((long)NB * NTOP1 * 3),   256, 0, stream>>>(pts, S.ord, pts1);
  };

  cloud(src, nsrc, xol, S.feat1s, S.pts1s, S.xols);
  cloud(tgt, ntgt, yol, S.feat1t, S.pts1t, S.sortv);   // tgt sorted scores unused

  // sim = fxa1 @ fya1^T   (M=N=1536, K=2560)
  gemm(S.feat1s, (long)NTOP1 * C5, 0, S.feat1t, (long)NTOP1 * C5, 1, nullptr, S.sim,
       (long)NTOP1 * NTOP1, NTOP1, NTOP1, C5);
  k_rowmax<<<dim3(NTOP1, NB), 256, 0, stream>>>(S.sim, S.smax);
  k_argsort<<<NB, 1024, 0, stream>>>(S.smax, S.smsort, S.smord, NTOP1, 2048);
  k_finalize<<<nblk((long)NB * NTOP2), 256, 0, stream>>>(S.sim, S.smord, S.pts1s,
      S.pts1t, S.xols, (float*)d_out);
}
